// SupConLossLambda_1726576853397
// MI455X (gfx1250) — matrix operand order repaired and checked by reading, not compiled, not run on hardware
//
#include <hip/hip_runtime.h>
#include <hip/hip_bf16.h>

typedef __attribute__((ext_vector_type(16))) _Float16 v16h;
typedef __attribute__((ext_vector_type(8)))  _Float16 v8h;
typedef __attribute__((ext_vector_type(4)))  _Float16 v4h;
typedef __attribute__((ext_vector_type(8)))  float    v8f;

#define LAMDA  0.5f
#define INV_T  (1.0f / 0.07f)
#define EPSV   1e-5f

// ---------------------------------------------------------------------------
// WMMA helper: D = A*B + C, f16 inputs, f32 accumulate (16x16x32)
// ---------------------------------------------------------------------------
__device__ __forceinline__ v8f wmma_f16(v16h a, v16h b, v8f c) {
  return __builtin_amdgcn_wmma_f32_16x16x32_f16(
      /*neg_a=*/false, a, /*neg_b=*/false, b,
      /*c_mod=*/(short)0, c, /*reuse_a=*/false, /*reuse_b=*/false);
}

// A operand (16x32, M x K), row-major f16 source with stride 128.
// ISA layout: lane m in [0,16) = row M=m holds K = 0..7 & 16..23,
//             lane m+16 = row M=m holds K = 8..15 & 24..31.
__device__ __forceinline__ v16h load_A(const _Float16* base, int row, int kc, int lane) {
  const _Float16* p = base + (size_t)row * 128 + kc * 32 + ((lane >> 4) & 1) * 8;
  v8h lo = *(const v8h*)(p);        // K = c0 .. c0+7
  v8h hi = *(const v8h*)(p + 16);   // K = c0+16 .. c0+23
  v16h r;
#pragma unroll
  for (int t = 0; t < 8; ++t) { r[t] = lo[t]; r[8 + t] = hi[t]; }
  return r;
}

// B operand (32x16, K x N), column N = lane&15 maps to source row `row`.
// Same K order as A: lanes 0-15 hold K=0..7 & 16..23, lanes 16-31 hold
// K=8..15 & 24..31.
__device__ __forceinline__ v16h load_B(const _Float16* base, int row, int kc, int lane) {
  const _Float16* p = base + (size_t)row * 128 + kc * 32 + ((lane >> 4) & 1) * 8;
  v8h lo = *(const v8h*)(p);
  v8h hi = *(const v8h*)(p + 16);
  v16h r;
#pragma unroll
  for (int t = 0; t < 8; ++t) { r[t] = lo[t]; r[8 + t] = hi[t]; }
  return r;
}

// Full 16-col B tile (hi+lo splits, all 4 K-chunks): 64 VGPRs
struct BTile {
  v16h h[4];
  v16h l[4];
};

__device__ __forceinline__ void load_Btile(BTile& b, const _Float16* __restrict__ fhi,
                                           const _Float16* __restrict__ flo,
                                           int jbase, int n15, int lane) {
#pragma unroll
  for (int kc = 0; kc < 4; ++kc) {
    b.h[kc] = load_B(fhi, jbase + n15, kc, lane);
    b.l[kc] = load_B(flo, jbase + n15, kc, lane);
  }
}

// 12 WMMAs: sim = hi*hi + lo*hi + hi*lo over K=128
__device__ __forceinline__ v8f compute_tile(const v16h ahi[4], const v16h alo[4],
                                            const BTile& b) {
  v8f c = {};
#pragma unroll
  for (int kc = 0; kc < 4; ++kc) {
    c = wmma_f16(ahi[kc], b.h[kc], c);
    c = wmma_f16(alo[kc], b.h[kc], c);
    c = wmma_f16(ahi[kc], b.l[kc], c);
  }
  return c;
}

// ---------------------------------------------------------------------------
// Pass 0: row L2-normalize; emit f16 hi + f16 lo (compensated split)
// ---------------------------------------------------------------------------
__global__ void __launch_bounds__(128)
normalize_split_kernel(const float* __restrict__ x,
                       _Float16* __restrict__ fhi, _Float16* __restrict__ flo, int N) {
  const int row  = blockIdx.x * 4 + (threadIdx.x >> 5);
  const int lane = threadIdx.x & 31;
  if (row >= N) return;
  const float4 v = *(const float4*)(x + (size_t)row * 128 + lane * 4);
  float ss = v.x * v.x + v.y * v.y + v.z * v.z + v.w * v.w;
#pragma unroll
  for (int off = 1; off < 32; off <<= 1) ss += __shfl_xor(ss, off, 32);
  const float inv = 1.0f / fmaxf(sqrtf(ss), 1e-12f);
  float f[4] = { v.x * inv, v.y * inv, v.z * inv, v.w * inv };
  v4h hh, hl;
#pragma unroll
  for (int t = 0; t < 4; ++t) {
    _Float16 h = (_Float16)f[t];
    hh[t] = h;
    hl[t] = (_Float16)(f[t] - (float)h);
  }
  *(v4h*)(fhi + (size_t)row * 128 + lane * 4) = hh;
  *(v4h*)(flo + (size_t)row * 128 + lane * 4) = hl;
}

// ---------------------------------------------------------------------------
// Pass 1: denom[i] = sum_{j != i} w_ij * exp(sim_ij) + EPS
// One block (4 waves) per 16-row anchor block; waves stride column tiles.
// B tiles double-buffered so WMMA never stalls on fresh loads.
// ---------------------------------------------------------------------------
__device__ __forceinline__ void denom_epilogue(v8f c, int jbase, int ibase, int n15,
                                               int hg, const int domi[8],
                                               const int* __restrict__ dom,
                                               float dsum[8]) {
  const int j    = jbase + n15;
  const int domj = dom[j];
#pragma unroll
  for (int v = 0; v < 8; ++v) {
    const int i    = ibase + v + 8 * hg;
    const float s  = c[v] * INV_T;
    const float w  = (domi[v] == domj) ? LAMDA : (1.0f - LAMDA);
    const float we = w * __expf(s);
    dsum[v] += (i == j) ? 0.0f : we;
  }
}

__global__ void __launch_bounds__(128)
pass_denom_kernel(const _Float16* __restrict__ fhi, const _Float16* __restrict__ flo,
                  const int* __restrict__ dom, float* __restrict__ denom, int N) {
  const int lane  = threadIdx.x & 31;
  const int wave  = threadIdx.x >> 5;
  const int n15   = lane & 15;
  const int hg    = (lane >> 4) & 1;
  const int ibase = blockIdx.x * 16;

  v16h ahi[4], alo[4];
#pragma unroll
  for (int kc = 0; kc < 4; ++kc) {
    ahi[kc] = load_A(fhi, ibase + n15, kc, lane);
    alo[kc] = load_A(flo, ibase + n15, kc, lane);
  }
  int domi[8];
#pragma unroll
  for (int v = 0; v < 8; ++v) domi[v] = dom[ibase + v + 8 * hg];

  float dsum[8];
#pragma unroll
  for (int v = 0; v < 8; ++v) dsum[v] = 0.0f;

  const int ntiles = N >> 4;          // assumes ntiles % 8 == 0 (N mult of 128)
  BTile b0, b1;
  load_Btile(b0, fhi, flo, wave << 4, n15, lane);
  for (int jt = wave; jt < ntiles; jt += 8) {
    const int jt2 = jt + 4;
    load_Btile(b1, fhi, flo, jt2 << 4, n15, lane);      // prefetch tile t+1
    v8f c0 = compute_tile(ahi, alo, b0);
    denom_epilogue(c0, jt << 4, ibase, n15, hg, domi, dom, dsum);
    if (jt + 8 < ntiles)
      load_Btile(b0, fhi, flo, (jt + 8) << 4, n15, lane); // prefetch tile t+2
    v8f c1 = compute_tile(ahi, alo, b1);
    denom_epilogue(c1, jt2 << 4, ibase, n15, hg, domi, dom, dsum);
  }

  // reduce each dsum[v] over its 16-lane half-group
#pragma unroll
  for (int v = 0; v < 8; ++v) {
    float x = dsum[v];
#pragma unroll
    for (int off = 1; off < 16; off <<= 1) x += __shfl_xor(x, off, 32);
    dsum[v] = x;
  }
  __shared__ float red[4][16];
#pragma unroll
  for (int v = 0; v < 8; ++v)
    if (n15 == v) red[wave][v + 8 * hg] = dsum[v];
  __syncthreads();
  if (threadIdx.x < 16) {
    const int t = threadIdx.x;
    denom[ibase + t] = red[0][t] + red[1][t] + red[2][t] + red[3][t] + EPSV;
  }
}

// ---------------------------------------------------------------------------
// Pass 2: per-row loss_i = sum_{pos} log(w e^s / denom_i + EPS) / max(npos,1)
// ---------------------------------------------------------------------------
__device__ __forceinline__ void loss_epilogue(v8f c, int jbase, int ibase, int n15,
                                              int hg, const int domi[8],
                                              const int labi[8], const float rden[8],
                                              const int* __restrict__ lab,
                                              const int* __restrict__ dom,
                                              float psum[8], float pcnt[8]) {
  const int j    = jbase + n15;
  const int labj = lab[j];
  const int domj = dom[j];
#pragma unroll
  for (int v = 0; v < 8; ++v) {
    const int i    = ibase + v + 8 * hg;
    const float s  = c[v] * INV_T;
    const float w  = (domi[v] == domj) ? LAMDA : (1.0f - LAMDA);
    const float we = w * __expf(s);
    const bool pos = (labi[v] == labj) && (i != j);
    const float lg = __logf(we * rden[v] + EPSV);
    psum[v] += pos ? lg : 0.0f;
    pcnt[v] += pos ? 1.0f : 0.0f;
  }
}

__global__ void __launch_bounds__(128)
pass_loss_kernel(const _Float16* __restrict__ fhi, const _Float16* __restrict__ flo,
                 const int* __restrict__ lab, const int* __restrict__ dom,
                 const float* __restrict__ denom,
                 float* __restrict__ lossv, float* __restrict__ nanv, int N) {
  const int lane  = threadIdx.x & 31;
  const int wave  = threadIdx.x >> 5;
  const int n15   = lane & 15;
  const int hg    = (lane >> 4) & 1;
  const int ibase = blockIdx.x * 16;

  v16h ahi[4], alo[4];
#pragma unroll
  for (int kc = 0; kc < 4; ++kc) {
    ahi[kc] = load_A(fhi, ibase + n15, kc, lane);
    alo[kc] = load_A(flo, ibase + n15, kc, lane);
  }
  int domi[8], labi[8];
  float rden[8];
#pragma unroll
  for (int v = 0; v < 8; ++v) {
    const int i = ibase + v + 8 * hg;
    domi[v] = dom[i];
    labi[v] = lab[i];
    rden[v] = 1.0f / denom[i];
  }

  float psum[8], pcnt[8];
#pragma unroll
  for (int v = 0; v < 8; ++v) { psum[v] = 0.0f; pcnt[v] = 0.0f; }

  const int ntiles = N >> 4;          // assumes ntiles % 8 == 0
  BTile b0, b1;
  load_Btile(b0, fhi, flo, wave << 4, n15, lane);
  for (int jt = wave; jt < ntiles; jt += 8) {
    const int jt2 = jt + 4;
    load_Btile(b1, fhi, flo, jt2 << 4, n15, lane);
    v8f c0 = compute_tile(ahi, alo, b0);
    loss_epilogue(c0, jt << 4, ibase, n15, hg, domi, labi, rden, lab, dom, psum, pcnt);
    if (jt + 8 < ntiles)
      load_Btile(b0, fhi, flo, (jt + 8) << 4, n15, lane);
    v8f c1 = compute_tile(ahi, alo, b1);
    loss_epilogue(c1, jt2 << 4, ibase, n15, hg, domi, labi, rden, lab, dom, psum, pcnt);
  }

#pragma unroll
  for (int v = 0; v < 8; ++v) {
    float a = psum[v], b = pcnt[v];
#pragma unroll
    for (int off = 1; off < 16; off <<= 1) {
      a += __shfl_xor(a, off, 32);
      b += __shfl_xor(b, off, 32);
    }
    psum[v] = a; pcnt[v] = b;
  }
  __shared__ float redS[4][16], redC[4][16];
#pragma unroll
  for (int v = 0; v < 8; ++v)
    if (n15 == v) { redS[wave][v + 8 * hg] = psum[v]; redC[wave][v + 8 * hg] = pcnt[v]; }
  __syncthreads();
  if (threadIdx.x < 16) {
    const int t = threadIdx.x;
    const float S  = redS[0][t] + redS[1][t] + redS[2][t] + redS[3][t];
    const float C  = redC[0][t] + redC[1][t] + redC[2][t] + redC[3][t];
    const float li = S / fmaxf(C, 1.0f);
    const bool valid = C > 0.5f;
    lossv[ibase + t] = valid ? li : 0.0f;
    nanv[ibase + t]  = valid ? 0.0f : 1.0f;
  }
}

// ---------------------------------------------------------------------------
// Final deterministic single-block reduction
// ---------------------------------------------------------------------------
__global__ void __launch_bounds__(256)
finalize_kernel(const float* __restrict__ lossv, const float* __restrict__ nanv,
                float* __restrict__ out, int N) {
  __shared__ float s1[256], s2[256];
  float a = 0.0f, b = 0.0f;
  for (int i = threadIdx.x; i < N; i += 256) { a += lossv[i]; b += nanv[i]; }
  s1[threadIdx.x] = a; s2[threadIdx.x] = b;
  __syncthreads();
  for (int off = 128; off > 0; off >>= 1) {
    if ((int)threadIdx.x < off) {
      s1[threadIdx.x] += s1[threadIdx.x + off];
      s2[threadIdx.x] += s2[threadIdx.x + off];
    }
    __syncthreads();
  }
  if (threadIdx.x == 0)
    out[0] = -s1[0] / ((float)N - s2[0] + 1.0f);
}

// ---------------------------------------------------------------------------
extern "C" void kernel_launch(void* const* d_in, const int* in_sizes, int n_in,
                              void* d_out, int out_size, void* d_ws, size_t ws_size,
                              hipStream_t stream) {
  const float* feat   = (const float*)d_in[0];
  const int*   labels = (const int*)d_in[1];
  const int*   doms   = (const int*)d_in[2];
  const int N = in_sizes[1];          // 8192
  // D assumed 128 (reference setup); layout below hardcodes stride 128.

  char* ws = (char*)d_ws;
  _Float16* fhi  = (_Float16*)(ws);
  _Float16* flo  = (_Float16*)(ws + (size_t)N * 128 * 2);
  float*    den  = (float*)   (ws + (size_t)N * 128 * 4);
  float*    lossv= (float*)   (ws + (size_t)N * 128 * 4 + (size_t)N * 4);
  float*    nanv = (float*)   (ws + (size_t)N * 128 * 4 + (size_t)N * 8);
  float*    out  = (float*)d_out;

  normalize_split_kernel<<<N / 4, 128, 0, stream>>>(feat, fhi, flo, N);
  pass_denom_kernel<<<N / 16, 128, 0, stream>>>(fhi, flo, doms, den, N);
  pass_loss_kernel<<<N / 16, 128, 0, stream>>>(fhi, flo, labels, doms, den, lossv, nanv, N);
  finalize_kernel<<<1, 256, 0, stream>>>(lossv, nanv, out, N);
}
